// Encoder_36902359007512
// MI455X (gfx1250) — compile-verified, matrix operand order checked by reading, not run on hardware
//
#include <hip/hip_runtime.h>
#include <cstdint>

typedef __attribute__((ext_vector_type(16))) _Float16 v16h;
typedef __attribute__((ext_vector_type(8)))  _Float16 v8h;
typedef __attribute__((ext_vector_type(8)))  float    v8f;

#define B_TOT   32768
#define TSTEPS  9
#define DIN     81
#define HID     128
#define NGATE   512          // 4*H
#define KDIM    209          // 81 + 128
#define KPAD    224          // 7 * 32 (WMMA K-chunks)
#define MROWS   32           // batch rows per workgroup
#define NTHR    256          // 8 waves (wave32)

__device__ __forceinline__ float sigmoidf_fast(float x) {
    return 1.0f / (1.0f + __expf(-x));
}

// Load one 16-lane-half WMMA fragment: 16 f16 per lane as two 16-byte LDS reads.
// Per CDNA5 16-bit A layout: lanes 0-15 hold K={k0..k0+7, k0+16..k0+23},
// lanes 16-31 hold the +8 variant (k0 already includes the per-lane-half +8).
__device__ __forceinline__ v16h ld_frag(const _Float16* rowbase, int k0) {
    v8h lo = *(const v8h*)(rowbase + k0);
    v8h hi = *(const v8h*)(rowbase + k0 + 16);
    v16h r;
#pragma unroll
    for (int i = 0; i < 8; ++i) { r[i] = lo[i]; r[i + 8] = hi[i]; }
    return r;
}

extern "C" __global__ __launch_bounds__(NTHR, 1)
void encoder_attn_lstm_kernel(const float* __restrict__ x,      // [B][9][81]
                              const float* __restrict__ W_attn, // [265]
                              const float* __restrict__ b_attn, // [1]
                              const float* __restrict__ W_ih,   // [512][81]
                              const float* __restrict__ W_hh,   // [512][128]
                              const float* __restrict__ b_ih,   // [512]
                              const float* __restrict__ b_hh,   // [512]
                              float* __restrict__ out_w,        // [B][9][81]
                              float* __restrict__ out_enc)      // [B][9][128]
{
    extern __shared__ __align__(16) char smem[];
    _Float16* Wlds = (_Float16*)smem;                 // [512][224] fused [W_ih | W_hh | 0]
    _Float16* Alds = Wlds + NGATE * KPAD;             // [32][224]  [w_in | h | 0]
    float*    Glds = (float*)(Alds + MROWS * KPAD);   // [32][512]  gate accumulators
    float*    Attn = Glds + MROWS * NGATE;            // [32][81]   softmax(pre)
    float*    Bias = Attn + MROWS * DIN;              // [512]      b_ih + b_hh

    const int tid    = threadIdx.x;
    const int wgbase = blockIdx.x * MROWS;

    // ---- one-time: fused weight matrix -> LDS as f16, row n = [W_ih[n,:81] | W_hh[n,:128] | 0]
    for (int p = tid; p < NGATE * KPAD; p += NTHR) {
        int n = p / KPAD, k = p - n * KPAD;
        float v = 0.0f;
        if (k < DIN)       v = W_ih[n * DIN + k];
        else if (k < KDIM) v = W_hh[n * HID + (k - DIN)];
        Wlds[p] = (_Float16)v;
    }
    // fused bias
    for (int p = tid; p < NGATE; p += NTHR) Bias[p] = b_ih[p] + b_hh[p];
    // zero h + pad region of the A tile (h0 = 0)
    for (int p = tid; p < MROWS * (KPAD - DIN); p += NTHR) {
        int r = p / (KPAD - DIN), k = DIN + (p - r * (KPAD - DIN));
        Alds[r * KPAD + k] = (_Float16)0.0f;
    }
    // pre[b,d] = sum_t x[b,t,d] * Wx[t] + b_attn
    const float battn = b_attn[0];
    for (int p = tid; p < MROWS * DIN; p += NTHR) {
        int r = p / DIN, d = p - r * DIN;
        const float* xp = x + ((size_t)(wgbase + r) * TSTEPS) * DIN + d;
        float acc = battn;
#pragma unroll
        for (int t = 0; t < TSTEPS; ++t) acc += xp[t * DIN] * W_attn[2 * HID + t];
        Attn[r * DIN + d] = acc;
    }
    __syncthreads();
    // softmax(pre) per row. softmax(pre + scalar) == softmax(pre), so the
    // reference's per-step (h.Wh + c.Wc) shift drops out: attn is step-invariant.
    if (tid < MROWS) {
        float* rowp = Attn + tid * DIN;
        float m = rowp[0];
        for (int d = 1; d < DIN; ++d) m = fmaxf(m, rowp[d]);
        float s = 0.0f;
        for (int d = 0; d < DIN; ++d) { float e = __expf(rowp[d] - m); rowp[d] = e; s += e; }
        float inv = 1.0f / s;
        for (int d = 0; d < DIN; ++d) rowp[d] *= inv;
    }
    __syncthreads();

    const int lane  = tid & 31;
    const int wv    = tid >> 5;          // wave 0..7 owns gate columns [wv*64, wv*64+64)
    const int lr    = lane & 15;         // row/col within 16-tile
    const int halfk = (lane >> 4) * 8;   // K sub-offset per 16-bit WMMA layout

    // pointwise-stage ownership: fixed hidden j, 16 rows (r0, r0+2, ...)
    const int jg = tid & 127;
    const int r0 = tid >> 7;
    const float bI = Bias[jg], bF = Bias[jg + 128], bG = Bias[jg + 256], bO = Bias[jg + 384];
    float c_state[16];
#pragma unroll
    for (int q = 0; q < 16; ++q) c_state[q] = 0.0f;

#pragma unroll 1
    for (int t = 0; t < TSTEPS; ++t) {
        // ---- stage 1: w_in = attn * x_t -> A tile (f16) + input_weighted (f32)
        for (int p = tid; p < MROWS * DIN; p += NTHR) {
            int r = p / DIN, d = p - r * DIN;
            size_t gb  = (size_t)(wgbase + r);
            size_t off = (gb * TSTEPS + t) * DIN + d;
            float wvv  = Attn[r * DIN + d] * x[off];
            out_w[off] = wvv;
            Alds[r * KPAD + d] = (_Float16)wvv;
        }
        __syncthreads();

        // ---- stage 2: gates(32x512) = A(32x224) * W^T via v_wmma_f32_16x16x32_f16
        v8f acc[2][4];
        {
            v8f z = {};
#pragma unroll
            for (int mt = 0; mt < 2; ++mt)
#pragma unroll
                for (int bt = 0; bt < 4; ++bt) acc[mt][bt] = z;
        }
#pragma unroll
        for (int kc = 0; kc < 7; ++kc) {
            const int k0 = kc * 32 + halfk;
            v16h a0 = ld_frag(Alds + (lr)      * KPAD, k0);
            v16h a1 = ld_frag(Alds + (16 + lr) * KPAD, k0);
#pragma unroll
            for (int bt = 0; bt < 4; ++bt) {
                const _Float16* wrow = Wlds + (wv * 64 + bt * 16 + lr) * KPAD;
                v16h bfr = ld_frag(wrow, k0);
                acc[0][bt] = __builtin_amdgcn_wmma_f32_16x16x32_f16(
                    false, a0, false, bfr, (short)0, acc[0][bt], false, false);
                acc[1][bt] = __builtin_amdgcn_wmma_f32_16x16x32_f16(
                    false, a1, false, bfr, (short)0, acc[1][bt], false, false);
            }
        }
        // scatter accumulators: C layout VGPR v -> M = v (+8 for upper lane half)
#pragma unroll
        for (int mt = 0; mt < 2; ++mt)
#pragma unroll
            for (int bt = 0; bt < 4; ++bt) {
                const int ncol = wv * 64 + bt * 16 + lr;
                const int mb   = mt * 16 + halfk;   // halfk == (lane>=16)*8
#pragma unroll
                for (int v = 0; v < 8; ++v)
                    Glds[(mb + v) * NGATE + ncol] = acc[mt][bt][v];
            }
        __syncthreads();

        // ---- stage 3: LSTM pointwise; c in registers, h2 -> A tile + input_encoded
#pragma unroll
        for (int q = 0; q < 16; ++q) {
            const int r = r0 + 2 * q;
            const float* gr = Glds + r * NGATE;
            float gi = gr[jg]        + bI;
            float gf = gr[jg + 128]  + bF;
            float gg = gr[jg + 256]  + bG;
            float go = gr[jg + 384]  + bO;
            float c2 = sigmoidf_fast(gf) * c_state[q] + sigmoidf_fast(gi) * tanhf(gg);
            float h2 = sigmoidf_fast(go) * tanhf(c2);
            c_state[q] = c2;
            Alds[r * KPAD + DIN + jg] = (_Float16)h2;
            out_enc[((size_t)(wgbase + r) * TSTEPS + t) * HID + jg] = h2;
        }
        // no trailing barrier needed: stage-1 writes of t+1 touch disjoint LDS,
        // and its __syncthreads() orders these h/c writes before the next GEMM.
    }
}

extern "C" void kernel_launch(void* const* d_in, const int* in_sizes, int n_in,
                              void* d_out, int out_size, void* d_ws, size_t ws_size,
                              hipStream_t stream) {
    const float* x      = (const float*)d_in[0];
    const float* W_attn = (const float*)d_in[1];
    const float* b_attn = (const float*)d_in[2];
    const float* W_ih   = (const float*)d_in[3];
    const float* W_hh   = (const float*)d_in[4];
    const float* b_ih   = (const float*)d_in[5];
    const float* b_hh   = (const float*)d_in[6];
    float* out_w   = (float*)d_out;
    float* out_enc = out_w + (size_t)B_TOT * TSTEPS * DIN;

    const size_t smem_bytes =
        (size_t)(NGATE * KPAD + MROWS * KPAD) * sizeof(_Float16) +
        (size_t)(MROWS * NGATE + MROWS * DIN + NGATE) * sizeof(float); // 321,664 B

    (void)hipFuncSetAttribute((const void*)encoder_attn_lstm_kernel,
                              hipFuncAttributeMaxDynamicSharedMemorySize,
                              (int)smem_bytes);

    encoder_attn_lstm_kernel<<<dim3(B_TOT / MROWS), dim3(NTHR), smem_bytes, stream>>>(
        x, W_attn, b_attn, W_ih, W_hh, b_ih, b_hh, out_w, out_enc);
}
